// Decoder_34385508172491
// MI455X (gfx1250) — matrix operand order repaired and checked by reading, compiled, not run on hardware
//
#include <hip/hip_runtime.h>
#include <hip/hip_bf16.h>

// ---------------------------------------------------------------------------
// Types for WMMA bf16 fragments
// ---------------------------------------------------------------------------
typedef __bf16 v16bf __attribute__((ext_vector_type(16)));
typedef float  v8f   __attribute__((ext_vector_type(8)));
typedef unsigned int v4u __attribute__((ext_vector_type(4)));

union FragU { v4u u[2]; v16bf v; };

#define WMMA_BF16(A, B, C) \
  __builtin_amdgcn_wmma_f32_16x16x32_bf16(false, (A), false, (B), (short)0, (C), false, false)

// Problem constants
#define VOCAB 32000
#define HDIM  1024
#define GDIM  4096     // 4*H
#define KCAT  2048     // E + H
#define BATCH 64
#define TSTEPS 127

// ---------------------------------------------------------------------------
// fp32 -> bf16 hi/lo split (round-to-nearest-even), pure bit ops
// ---------------------------------------------------------------------------
__device__ __forceinline__ unsigned short f32_to_bf16(float x) {
  unsigned int u = __float_as_uint(x);
  unsigned int r = u + 0x7FFFu + ((u >> 16) & 1u);
  return (unsigned short)(r >> 16);
}
__device__ __forceinline__ void split_bf16x2(float x, unsigned short* hi, unsigned short* lo) {
  unsigned short h = f32_to_bf16(x);
  float fh = __uint_as_float(((unsigned int)h) << 16);
  *hi = h;
  *lo = f32_to_bf16(x - fh);
}

// ---------------------------------------------------------------------------
// Fragment loaders. rowk = pointer to this lane's row, already offset by kb.
// A 16x32 bf16: lane L -> M = L%16, half = L/16; elems 0..7 : K = half*8+e,
//                                            elems 8..15: K = 16+half*8+(e-8)
// B 32x16 bf16: lane L -> N = L%16, kh = L/16; elems 0..7: K = kh*8+e, elems 8..15: K = 16+kh*8+(e-8)
// ---------------------------------------------------------------------------
__device__ __forceinline__ v16bf frag_a(const unsigned short* rowk, int lane) {
  int half = lane >> 4;
  FragU U;
  U.u[0] = *(const v4u*)(rowk + half * 8);
  U.u[1] = *(const v4u*)(rowk + 16 + half * 8);
  return U.v;
}
__device__ __forceinline__ v16bf frag_b(const unsigned short* rowk, int lane) {
  int kh = lane >> 4;
  FragU U;
  U.u[0] = *(const v4u*)(rowk + kh * 8);
  U.u[1] = *(const v4u*)(rowk + 16 + kh * 8);
  return U.v;
}

__device__ __forceinline__ float sigmoidf(float x) { return 1.0f / (1.0f + expf(-x)); }

// ---------------------------------------------------------------------------
// One-time prep kernels
// ---------------------------------------------------------------------------
__global__ void zero_loss_kernel(float* out) { out[0] = 0.0f; }

// Wcat[j][k] = k<H ? W_ih[j][k] : W_hh[j][k-H]   -> bf16 hi/lo planes (GDIM x KCAT)
__global__ void prep_wcat_kernel(const float* __restrict__ W_ih,
                                 const float* __restrict__ W_hh,
                                 unsigned short* __restrict__ Whi,
                                 unsigned short* __restrict__ Wlo) {
  int i = blockIdx.x * blockDim.x + threadIdx.x;
  if (i >= GDIM * KCAT) return;
  int j = i >> 11, k = i & (KCAT - 1);
  float v = (k < HDIM) ? W_ih[(size_t)j * HDIM + k] : W_hh[(size_t)j * HDIM + (k - HDIM)];
  split_bf16x2(v, &Whi[i], &Wlo[i]);
}

// W_out (VOCAB x HDIM) -> bf16 hi/lo planes (already [n][k] for B-matrix)
__global__ void prep_wout_kernel(const float* __restrict__ W_out,
                                 unsigned short* __restrict__ Whi,
                                 unsigned short* __restrict__ Wlo) {
  int i = blockIdx.x * blockDim.x + threadIdx.x;
  if (i >= VOCAB * HDIM) return;
  split_bf16x2(W_out[i], &Whi[i], &Wlo[i]);
}

__global__ void prep_bias_kernel(const float* __restrict__ b_ih,
                                 const float* __restrict__ b_hh,
                                 float* __restrict__ bias) {
  int i = blockIdx.x * blockDim.x + threadIdx.x;
  if (i < GDIM) bias[i] = b_ih[i] + b_hh[i];
}

// X[t][b][k] = (tok==0) ? 0 : embedding[tok][k], packed to bf16 hi/lo planes
__global__ void prep_x_kernel(const int* __restrict__ tokens,
                              const float* __restrict__ emb,
                              unsigned short* __restrict__ Xhi,
                              unsigned short* __restrict__ Xlo) {
  int i = blockIdx.x * blockDim.x + threadIdx.x;
  if (i >= TSTEPS * BATCH * HDIM) return;
  int k  = i & (HDIM - 1);
  int tb = i >> 10;           // t*64 + b
  int b  = tb & (BATCH - 1);
  int t  = tb >> 6;
  int tok = tokens[t * BATCH + b];
  float v = (tok == 0) ? 0.0f : emb[(size_t)tok * HDIM + k];
  split_bf16x2(v, &Xhi[i], &Xlo[i]);
}

// copy h0/c0 into workspace state; pack h into bf16 hi/lo planes
__global__ void init_state_kernel(const float* __restrict__ h0,
                                  const float* __restrict__ c0,
                                  float* __restrict__ h, float* __restrict__ c,
                                  unsigned short* __restrict__ Hhi,
                                  unsigned short* __restrict__ Hlo) {
  int i = blockIdx.x * blockDim.x + threadIdx.x;
  if (i >= BATCH * HDIM) return;
  float hv = h0[i];
  h[i] = hv; c[i] = c0[i];
  split_bf16x2(hv, &Hhi[i], &Hlo[i]);
}

// ---------------------------------------------------------------------------
// Core bf16x3 multiply-accumulate over one 32-K chunk for 4 N-tiles,
// interleaved so consecutive WMMAs hit different accumulators (hides the
// bf16 WMMA->WMMA hazard window with independent WMMAs instead of v_nops).
// ---------------------------------------------------------------------------
__device__ __forceinline__ void mma_chunk4(v8f acc[4], v16bf ah, v16bf al,
                                           const unsigned short* const pbh[4],
                                           const unsigned short* const pbl[4],
                                           int kb, int lane) {
  v16bf bh[4], bl[4];
  #pragma unroll
  for (int j = 0; j < 4; ++j) {
    __builtin_prefetch((const void*)(pbh[j] + kb + 32), 0, 3);
    bh[j] = frag_b(pbh[j] + kb, lane);
    bl[j] = frag_b(pbl[j] + kb, lane);
  }
  #pragma unroll
  for (int j = 0; j < 4; ++j) acc[j] = WMMA_BF16(ah, bh[j], acc[j]);  // hi*hi
  #pragma unroll
  for (int j = 0; j < 4; ++j) acc[j] = WMMA_BF16(ah, bl[j], acc[j]);  // hi*lo
  #pragma unroll
  for (int j = 0; j < 4; ++j) acc[j] = WMMA_BF16(al, bh[j], acc[j]);  // lo*hi
}

// ---------------------------------------------------------------------------
// gates = [X_t | h] @ Wcat^T + bias    (M=64, N=4096, K=2048)  bf16x3 WMMA
// Each wave computes 16(M) x 64(N): 4 interleaved accumulators.
// 256 waves -> 32 blocks of 8 waves. Two branch-free K loops (X half, H half).
// ---------------------------------------------------------------------------
__global__ void gates_gemm_kernel(const unsigned short* __restrict__ Xhi,
                                  const unsigned short* __restrict__ Xlo,
                                  const unsigned short* __restrict__ Hhi,
                                  const unsigned short* __restrict__ Hlo,
                                  const unsigned short* __restrict__ Whi,
                                  const unsigned short* __restrict__ Wlo,
                                  const float* __restrict__ bias,
                                  float* __restrict__ gates, int t) {
  int wave = threadIdx.x >> 5;
  int lane = threadIdx.x & 31;
  int tile = blockIdx.x * 8 + wave;        // 0..255
  int mt = tile & 3;                        // batch tile
  int ng = tile >> 2;                       // 0..63 : group of 4 n-tiles
  int mbase = mt * 16;
  int m = mbase + (lane & 15);
  int nrow0 = ng * 64 + (lane & 15);        // B row for j-th n-tile: nrow0 + j*16

  const unsigned short* pbh[4];
  const unsigned short* pbl[4];
  #pragma unroll
  for (int j = 0; j < 4; ++j) {
    size_t ro = (size_t)(nrow0 + j * 16) * KCAT;
    pbh[j] = Whi + ro;
    pbl[j] = Wlo + ro;
  }

  v8f acc[4] = {};
  // ---- K in [0, HDIM): A from X_t ----
  const unsigned short* axh = Xhi + ((size_t)t * BATCH + m) * HDIM;
  const unsigned short* axl = Xlo + ((size_t)t * BATCH + m) * HDIM;
  for (int kb = 0; kb < HDIM; kb += 32) {
    v16bf ah = frag_a(axh + kb, lane);
    v16bf al = frag_a(axl + kb, lane);
    mma_chunk4(acc, ah, al, pbh, pbl, kb, lane);
  }
  // ---- K in [HDIM, KCAT): A from h ----
  const unsigned short* ahh = Hhi + (size_t)m * HDIM;
  const unsigned short* ahl = Hlo + (size_t)m * HDIM;
  for (int kb = 0; kb < HDIM; kb += 32) {
    v16bf ah = frag_a(ahh + kb, lane);
    v16bf al = frag_a(ahl + kb, lane);
    mma_chunk4(acc, ah, al, pbh, pbl, HDIM + kb, lane);
  }

  // C layout: lane -> N=lane%16, mhalf=lane/16; VGPR r -> M=mhalf*8+r
  int mh = lane >> 4;
  #pragma unroll
  for (int j = 0; j < 4; ++j) {
    int n = ng * 64 + j * 16 + (lane & 15);
    float bj = bias[n];
    #pragma unroll
    for (int r = 0; r < 8; ++r) {
      int b = mbase + mh * 8 + r;
      gates[(size_t)b * GDIM + n] = acc[j][r] + bj;
    }
  }
}

// ---------------------------------------------------------------------------
// LSTM elementwise update; also re-packs new h into bf16 hi/lo planes
// ---------------------------------------------------------------------------
__global__ void lstm_update_kernel(const float* __restrict__ gates,
                                   float* __restrict__ h, float* __restrict__ c,
                                   unsigned short* __restrict__ Hhi,
                                   unsigned short* __restrict__ Hlo) {
  int i = blockIdx.x * blockDim.x + threadIdx.x;
  if (i >= BATCH * HDIM) return;
  int b = i >> 10, k = i & (HDIM - 1);
  const float* g = gates + (size_t)b * GDIM;
  float gi = g[k], gf = g[HDIM + k], gg = g[2 * HDIM + k], go = g[3 * HDIM + k];
  float cn = sigmoidf(gf) * c[i] + sigmoidf(gi) * tanhf(gg);
  float hn = sigmoidf(go) * tanhf(cn);
  c[i] = cn; h[i] = hn;
  split_bf16x2(hn, &Hhi[i], &Hlo[i]);
}

// ---------------------------------------------------------------------------
// dist = h @ W_out^T + b_out    (M=64, N=32000, K=1024)  bf16x3 WMMA
// Each wave computes 16(M) x 64(N): 4 interleaved accumulators.
// 2000 waves -> 250 blocks of 8 waves.
// ---------------------------------------------------------------------------
__global__ void logits_gemm_kernel(const unsigned short* __restrict__ Hhi,
                                   const unsigned short* __restrict__ Hlo,
                                   const unsigned short* __restrict__ Whi,
                                   const unsigned short* __restrict__ Wlo,
                                   const float* __restrict__ b_out,
                                   float* __restrict__ dist) {
  int wave = threadIdx.x >> 5;
  int lane = threadIdx.x & 31;
  int tile = blockIdx.x * 8 + wave;        // 0..1999
  int mt = tile & 3;
  int ng = tile >> 2;                      // 0..499
  int mbase = mt * 16;
  int m = mbase + (lane & 15);
  int nrow0 = ng * 64 + (lane & 15);

  const unsigned short* pbh[4];
  const unsigned short* pbl[4];
  #pragma unroll
  for (int j = 0; j < 4; ++j) {
    size_t ro = (size_t)(nrow0 + j * 16) * HDIM;
    pbh[j] = Whi + ro;
    pbl[j] = Wlo + ro;
  }

  const unsigned short* arh = Hhi + (size_t)m * HDIM;
  const unsigned short* arl = Hlo + (size_t)m * HDIM;

  v8f acc[4] = {};
  for (int kb = 0; kb < HDIM; kb += 32) {
    v16bf ah = frag_a(arh + kb, lane);
    v16bf al = frag_a(arl + kb, lane);
    mma_chunk4(acc, ah, al, pbh, pbl, kb, lane);
  }

  int mh = lane >> 4;
  #pragma unroll
  for (int j = 0; j < 4; ++j) {
    int n = ng * 64 + j * 16 + (lane & 15);
    float bn = b_out[n];
    #pragma unroll
    for (int r = 0; r < 8; ++r) {
      int b = mbase + mh * 8 + r;
      dist[(size_t)b * VOCAB + n] = acc[j][r] + bn;
    }
  }
}

// ---------------------------------------------------------------------------
// Per-row (batch) reduction over 32000 logits: max, sum-exp, first-argmax,
// label NLL. One block per row.
// ---------------------------------------------------------------------------
__global__ void row_reduce_kernel(const float* __restrict__ dist,
                                  const int* __restrict__ tokens,
                                  int t, float* __restrict__ stats) {
  int b = blockIdx.x;
  const float* row = dist + (size_t)b * VOCAB;
  __shared__ float smax[256];
  __shared__ float ssum[256];
  __shared__ int   sidx[256];
  int tid = threadIdx.x;

  // phase 1: row max
  float mx = -3.402823466e38f;
  for (int n = tid; n < VOCAB; n += 256) mx = fmaxf(mx, row[n]);
  smax[tid] = mx;
  __syncthreads();
  for (int s = 128; s > 0; s >>= 1) {
    if (tid < s) smax[tid] = fmaxf(smax[tid], smax[tid + s]);
    __syncthreads();
  }
  float rowmax = smax[0];
  __syncthreads();

  // phase 2: sum-exp + first-occurrence argmax
  float lsum = 0.0f;
  float bv = -3.402823466e38f;
  int bi = 0x7fffffff;
  for (int n = tid; n < VOCAB; n += 256) {
    float x = row[n];
    lsum += expf(x - rowmax);
    if (x > bv || (x == bv && n < bi)) { bv = x; bi = n; }
  }
  ssum[tid] = lsum; smax[tid] = bv; sidx[tid] = bi;
  __syncthreads();
  for (int s = 128; s > 0; s >>= 1) {
    if (tid < s) {
      ssum[tid] += ssum[tid + s];
      float ov = smax[tid + s]; int oi = sidx[tid + s];
      if (ov > smax[tid] || (ov == smax[tid] && oi < sidx[tid])) {
        smax[tid] = ov; sidx[tid] = oi;
      }
    }
    __syncthreads();
  }

  if (tid == 0) {
    int label = tokens[(t + 1) * BATCH + b];
    float logit = row[label];
    float nll = -(logit - rowmax - logf(ssum[0]));
    float valid = (label != 0) ? 1.0f : 0.0f;
    stats[b * 4 + 0] = nll * valid;
    stats[b * 4 + 1] = valid;
    stats[b * 4 + 2] = (float)sidx[0];
  }
}

// ---------------------------------------------------------------------------
// Per-step loss accumulate + write predictions (argmax) as floats
// ---------------------------------------------------------------------------
__global__ void loss_step_kernel(const float* __restrict__ stats, int t,
                                 float* __restrict__ out) {
  __shared__ float snll[64];
  __shared__ float sval[64];
  int b = threadIdx.x;      // 64 threads
  snll[b] = stats[b * 4 + 0];
  sval[b] = stats[b * 4 + 1];
  __syncthreads();
  for (int s = 32; s > 0; s >>= 1) {
    if (b < s) { snll[b] += snll[b + s]; sval[b] += sval[b + s]; }
    __syncthreads();
  }
  if (b == 0) out[0] += snll[0] / fmaxf(sval[0], 1.0f);
  out[1 + t * BATCH + b] = stats[b * 4 + 2];
}

// ---------------------------------------------------------------------------
// Launcher
// ---------------------------------------------------------------------------
extern "C" void kernel_launch(void* const* d_in, const int* in_sizes, int n_in,
                              void* d_out, int out_size, void* d_ws, size_t ws_size,
                              hipStream_t stream) {
  const int*   tokens = (const int*)d_in[0];
  const float* h0     = (const float*)d_in[4];
  const float* c0     = (const float*)d_in[5];
  const float* embed  = (const float*)d_in[6];
  const float* W_ih   = (const float*)d_in[7];
  const float* W_hh   = (const float*)d_in[8];
  const float* b_ih   = (const float*)d_in[9];
  const float* b_hh   = (const float*)d_in[10];
  const float* W_out  = (const float*)d_in[11];
  const float* b_out  = (const float*)d_in[12];
  float* out = (float*)d_out;

  char* w = (char*)d_ws;
  size_t cur = 0;
  auto take = [&](size_t bytes) -> void* {
    cur = (cur + 255) & ~(size_t)255;
    void* p = (void*)(w + cur);
    cur += bytes;
    return p;
  };
  unsigned short* WcatHi = (unsigned short*)take((size_t)GDIM * KCAT * 2);
  unsigned short* WcatLo = (unsigned short*)take((size_t)GDIM * KCAT * 2);
  unsigned short* WoutHi = (unsigned short*)take((size_t)VOCAB * HDIM * 2);
  unsigned short* WoutLo = (unsigned short*)take((size_t)VOCAB * HDIM * 2);
  float*          bias   = (float*)take((size_t)GDIM * 4);
  unsigned short* Xhi    = (unsigned short*)take((size_t)TSTEPS * BATCH * HDIM * 2);
  unsigned short* Xlo    = (unsigned short*)take((size_t)TSTEPS * BATCH * HDIM * 2);
  float*          hbuf   = (float*)take((size_t)BATCH * HDIM * 4);
  float*          cbuf   = (float*)take((size_t)BATCH * HDIM * 4);
  unsigned short* Hhi    = (unsigned short*)take((size_t)BATCH * HDIM * 2);
  unsigned short* Hlo    = (unsigned short*)take((size_t)BATCH * HDIM * 2);
  float*          gates  = (float*)take((size_t)BATCH * GDIM * 4);
  float*          dist   = (float*)take((size_t)BATCH * VOCAB * 4);
  float*          stats  = (float*)take((size_t)BATCH * 4 * 4);

  zero_loss_kernel<<<1, 1, 0, stream>>>(out);
  prep_wcat_kernel<<<(GDIM * KCAT) / 256, 256, 0, stream>>>(W_ih, W_hh, WcatHi, WcatLo);
  prep_wout_kernel<<<(VOCAB * HDIM) / 256, 256, 0, stream>>>(W_out, WoutHi, WoutLo);
  prep_bias_kernel<<<GDIM / 256, 256, 0, stream>>>(b_ih, b_hh, bias);
  prep_x_kernel<<<(TSTEPS * BATCH * HDIM) / 256, 256, 0, stream>>>(tokens, embed, Xhi, Xlo);
  init_state_kernel<<<(BATCH * HDIM) / 256, 256, 0, stream>>>(h0, c0, hbuf, cbuf, Hhi, Hlo);

  for (int t = 0; t < TSTEPS; ++t) {
    gates_gemm_kernel<<<32, 256, 0, stream>>>(Xhi, Xlo, Hhi, Hlo, WcatHi, WcatLo, bias, gates, t);
    lstm_update_kernel<<<(BATCH * HDIM) / 256, 256, 0, stream>>>(gates, hbuf, cbuf, Hhi, Hlo);
    logits_gemm_kernel<<<250, 256, 0, stream>>>(Hhi, Hlo, WoutHi, WoutLo, b_out, dist);
    row_reduce_kernel<<<BATCH, 256, 0, stream>>>(dist, tokens, t, stats);
    loss_step_kernel<<<1, 64, 0, stream>>>(stats, t, out);
  }
}
